// FuzzyRNN_23794118820467
// MI455X (gfx1250) — matrix operand order repaired and checked by reading, not compiled, not run on hardware
//
#include <hip/hip_runtime.h>

typedef __attribute__((ext_vector_type(16))) __bf16 v16bf;
typedef __attribute__((ext_vector_type(8)))  float  v8f;

union V16 { v16bf v; uint4 u[2]; };

#define Bsz 512
#define Tsz 512
#define Dsz 128
#define Hsz 256
#define KF  3
#define EPSF 1e-8f

__device__ __forceinline__ float fast_rcp(float x) { return __builtin_amdgcn_rcpf(x); }
// tanh(x) = 1 - 2/(exp(2x)+1); correct limits at +/-inf via exp saturation
__device__ __forceinline__ float fast_tanh(float x) {
  float e = __expf(2.0f * x);
  return 1.0f - 2.0f * fast_rcp(e + 1.0f);
}
__device__ __forceinline__ float fast_sigmoid(float x) {
  return fast_rcp(1.0f + __expf(-x));
}

// ---------------------------------------------------------------------------
// Phase 1: xW[B*T, H] = x[B*T, D] @ Wx[D, H]   (bf16 WMMA, f32 accumulate)
// grid = (B*T/16), block = 256: 8 waves, each one 16x32 tile (A reused 2x)
// ---------------------------------------------------------------------------
__global__ __launch_bounds__(256)
void fuzzyrnn_xw_gemm(const float* __restrict__ x, const float* __restrict__ Wx,
                      float* __restrict__ xw) {
  const int wave = threadIdx.x >> 5;
  const int lane = threadIdx.x & 31;
  const int l16  = lane & 15;
  const int half = lane >> 4;
  const int m0 = blockIdx.x * 16;
  const int n0 = wave * 32;

  v8f acc0 = {}, acc1 = {};
  const float* arow = x + (size_t)(m0 + l16) * Dsz;
#pragma unroll
  for (int kc = 0; kc < 4; ++kc) {
    V16 a, b0, b1;
    // A: 16x32 bf16, lane = row M, elements sequential in K, halves at K=16
#pragma unroll
    for (int e = 0; e < 8; ++e) a.v[e]     = (__bf16)arow[kc * 32 + half * 16 + e];
#pragma unroll
    for (int e = 0; e < 8; ++e) a.v[8 + e] = (__bf16)arow[kc * 32 + half * 16 + 8 + e];
    // B: 32x16 bf16, lane = col N, elements sequential in K, halves at K=16
#pragma unroll
    for (int e = 0; e < 16; ++e) {
      int k = kc * 32 + half * 16 + e;
      b0.v[e] = (__bf16)Wx[k * Hsz + n0 + l16];
      b1.v[e] = (__bf16)Wx[k * Hsz + n0 + 16 + l16];
    }
    acc0 = __builtin_amdgcn_wmma_f32_16x16x32_bf16(false, a.v, false, b0.v,
                                                   (short)0, acc0, false, false);
    acc1 = __builtin_amdgcn_wmma_f32_16x16x32_bf16(false, a.v, false, b1.v,
                                                   (short)0, acc1, false, false);
  }
#pragma unroll
  for (int r = 0; r < 8; ++r) {
    int m = m0 + half * 8 + r;
    xw[(size_t)m * Hsz + n0 + l16]      = acc0[r];
    xw[(size_t)m * Hsz + n0 + 16 + l16] = acc1[r];
  }
}

// ---------------------------------------------------------------------------
// Phase 2: recurrent scan. One WG per 16 batch rows (grid = 32, block = 256).
// Wh in registers as WMMA B-fragments; h state f32 in registers; bf16 mirror
// of h double-buffered in LDS (single barrier per timestep).
// ---------------------------------------------------------------------------
__global__ __launch_bounds__(256)
void fuzzyrnn_scan(const float* __restrict__ xw, const float* __restrict__ Wh,
                   const float* __restrict__ bvec, const float* __restrict__ cC,
                   const float* __restrict__ sg, const float* __restrict__ qC,
                   const float* __restrict__ Wc, const float* __restrict__ bc,
                   float* __restrict__ out) {
  __shared__ __align__(16) __bf16 hbf[2][16 * Hsz];   // 16 KB ping-pong
  __shared__ float accum[16];

  const int wave = threadIdx.x >> 5;
  const int lane = threadIdx.x & 31;
  const int l16  = lane & 15;
  const int half = lane >> 4;
  const int bblk = blockIdx.x * 16;   // 16 batch rows owned by this WG
  const int n0   = wave * 32;         // 32 H-columns owned by this wave

  // ---- preload Wh as bf16 WMMA B fragments (constant over time) ----------
  V16 bfr[2][8];
#pragma unroll
  for (int tc = 0; tc < 2; ++tc) {
    int n = n0 + tc * 16 + l16;
#pragma unroll
    for (int kc = 0; kc < 8; ++kc) {
#pragma unroll
      for (int e = 0; e < 16; ++e) {
        int k = kc * 32 + half * 16 + e;
        bfr[tc][kc].v[e] = (__bf16)Wh[k * Hsz + n];
      }
    }
  }

  // ---- per-lane gate coefficients (each lane owns 2 columns j) -----------
  int jj[2] = { n0 + l16, n0 + 16 + l16 };
  float bb[2], wc[2], cc[2][KF], is2[2][KF], qq[2][KF];
#pragma unroll
  for (int tc = 0; tc < 2; ++tc) {
    bb[tc] = bvec[jj[tc]];
    wc[tc] = Wc[jj[tc]];
#pragma unroll
    for (int k = 0; k < KF; ++k) {
      cc[tc][k] = cC[jj[tc] * KF + k];
      float s   = sg[jj[tc] * KF + k];
      is2[tc][k] = 1.0f / (2.0f * s * s + EPSF);
      qq[tc][k] = qC[jj[tc] * KF + k];
    }
  }

  float hcur[16];
#pragma unroll
  for (int i = 0; i < 16; ++i) hcur[i] = 0.0f;
#pragma unroll
  for (int i = 0; i < 16; ++i) hbf[0][threadIdx.x * 16 + i] = (__bf16)0.0f;
  if (threadIdx.x < 16) accum[threadIdx.x] = 0.0f;
  __syncthreads();

  int c = 0;
  for (int t = 0; t < Tsz; ++t) {
    // 1) issue global xw loads early (independent of h) + prefetch t+1
    float xv[16];
#pragma unroll
    for (int tc = 0; tc < 2; ++tc) {
#pragma unroll
      for (int r = 0; r < 8; ++r) {
        int m = half * 8 + r;
        int j = n0 + tc * 16 + l16;
        const size_t xidx = ((size_t)(bblk + m) * Tsz + t) * Hsz + j;
        xv[tc * 8 + r] = xw[xidx];
        if (tc == 0 && t + 1 < Tsz)
          __builtin_prefetch(&xw[xidx + Hsz], 0, 0);
      }
    }

    // 2) A fragments of h from LDS buf[c]: two b128 loads per K-chunk
    V16 afr[8];
    const char* hb = (const char*)(&hbf[c][0]) + l16 * (Hsz * 2);
#pragma unroll
    for (int kc = 0; kc < 8; ++kc) {
      afr[kc].u[0] = *(const uint4*)(hb + kc * 64 + half * 32);
      afr[kc].u[1] = *(const uint4*)(hb + kc * 64 + half * 32 + 16);
    }

    // 3) h @ Wh via bf16 WMMA (two interleaved 8-deep chains)
    v8f c0 = {}, c1 = {};
#pragma unroll
    for (int kc = 0; kc < 8; ++kc) {
      c0 = __builtin_amdgcn_wmma_f32_16x16x32_bf16(false, afr[kc].v, false,
                                                   bfr[0][kc].v, (short)0, c0,
                                                   false, false);
      c1 = __builtin_amdgcn_wmma_f32_16x16x32_bf16(false, afr[kc].v, false,
                                                   bfr[1][kc].v, (short)0, c1,
                                                   false, false);
    }

    // 4) fuzzy gate + state update -> LDS buf[1-c]
    __bf16* hn_out = &hbf[1 - c][0];
#pragma unroll
    for (int tc = 0; tc < 2; ++tc) {
#pragma unroll
      for (int r = 0; r < 8; ++r) {
        int m = half * 8 + r;
        int j = n0 + tc * 16 + l16;
        float xwv = xv[tc * 8 + r];
        float z = xwv + (tc ? c1[r] : c0[r]) + bb[tc];
        float num = 0.0f, den = 0.0f;
#pragma unroll
        for (int k = 0; k < KF; ++k) {
          float d  = z - cc[tc][k];
          float mu = __expf(-d * d * is2[tc][k]);
          num += mu * qq[tc][k];
          den += mu;
        }
        float g  = fast_sigmoid(num * fast_rcp(den + EPSF));
        int idx  = tc * 8 + r;
        float hn = (1.0f - g) * hcur[idx] + g * fast_tanh(xwv);
        hcur[idx] = hn;
        hn_out[m * Hsz + j] = (__bf16)hn;
      }
    }
    __syncthreads();   // writes to buf[1-c] visible; reads of buf[c] done
    c ^= 1;
  }

  // ---- out[b] = h_final[b,:] @ Wc + bc ------------------------------------
#pragma unroll
  for (int tc = 0; tc < 2; ++tc)
#pragma unroll
    for (int r = 0; r < 8; ++r)
      atomicAdd(&accum[half * 8 + r], hcur[tc * 8 + r] * wc[tc]);
  __syncthreads();
  if (threadIdx.x < 16) out[bblk + threadIdx.x] = accum[threadIdx.x] + bc[0];
}

// ---------------------------------------------------------------------------
extern "C" void kernel_launch(void* const* d_in, const int* in_sizes, int n_in,
                              void* d_out, int out_size, void* d_ws, size_t ws_size,
                              hipStream_t stream) {
  (void)in_sizes; (void)n_in; (void)out_size; (void)ws_size;
  const float* x  = (const float*)d_in[0];
  const float* Wx = (const float*)d_in[1];
  const float* Wh = (const float*)d_in[2];
  const float* bv = (const float*)d_in[3];
  const float* cC = (const float*)d_in[4];
  const float* sg = (const float*)d_in[5];
  const float* qC = (const float*)d_in[6];
  const float* Wc = (const float*)d_in[7];
  const float* bc = (const float*)d_in[8];
  float* out = (float*)d_out;
  float* xw  = (float*)d_ws;   // staging: B*T*H floats = 256 MiB

  fuzzyrnn_xw_gemm<<<(Bsz * Tsz) / 16, 256, 0, stream>>>(x, Wx, xw);
  fuzzyrnn_scan<<<Bsz / 16, 256, 0, stream>>>(xw, Wh, bv, cC, sg, qC, Wc, bc, out);
}
